// NonLocalGeometricBlock_49168785605359
// MI455X (gfx1250) — matrix operand order repaired and checked by reading, not compiled, not run on hardware
//
#include <hip/hip_runtime.h>
#include <stdint.h>

#define N_PTS 32768
#define DIM   512
#define MF    512
#define KEPS  0.001f
#define NSTAB 1e-6f
#define KV_CHUNKS 8

typedef __attribute__((ext_vector_type(16))) __bf16 v16bf;
typedef __attribute__((ext_vector_type(8)))  float  v8f;
typedef __attribute__((ext_vector_type(4)))  float  v4f;

__device__ __forceinline__ unsigned short f2bf(float f) {
  union { float f; uint32_t u; } c; c.f = f;
  uint32_t u = c.u + 0x7FFFu + ((c.u >> 16) & 1u);   // round-to-nearest-even
  return (unsigned short)(u >> 16);
}
__device__ __forceinline__ float bf2f(unsigned short h) {
  union { uint32_t u; float f; } c; c.u = ((uint32_t)h) << 16;
  return c.f;
}

union VBF { v16bf v; unsigned short u[16]; };

// ---- WMMA tile loaders (contiguous k = 16h + s for both A and B, wave32) ----

// A tile 16(M)x32(K) bf16 from row-major bf16 src.
// lanes 0-15: row M=lane, K = k0..k0+15
// lanes 16-31: row M=lane-16, K = k0+16..k0+31
__device__ __forceinline__ v16bf loadA_bf(const unsigned short* src, int ld, int row0, int k0) {
  int lane = threadIdx.x & 31;
  int r    = row0 + (lane & 15);
  int kb   = k0 + ((lane >> 4) << 4);
  const unsigned short* p1 = src + (size_t)r * ld + kb;
  VBF o;
#pragma unroll
  for (int j = 0; j < 16; ++j) o.u[j] = p1[j];
  return o.v;
}

// B tile 32(K)x16(N) from a matrix stored TRANSPOSED row-major (row = output col).
// lanes 0-15: col N=lane, K = k0..k0+15 ; lanes 16-31: col N=lane-16, K = k0+16..k0+31
__device__ __forceinline__ v16bf loadBT_bf(const unsigned short* srcT, int ld, int col0, int k0) {
  int lane = threadIdx.x & 31;
  int c    = col0 + (lane & 15);
  int kb   = k0 + ((lane >> 4) << 4);
  const unsigned short* p = srcT + (size_t)c * ld + kb;
  VBF o;
#pragma unroll
  for (int j = 0; j < 16; ++j) o.u[j] = p[j];
  return o.v;
}

#define WMMA_BF16(a, b, c) \
  __builtin_amdgcn_wmma_f32_16x16x32_bf16(false, (a), false, (b), (short)0, (c), false, false)

// ---- Kernel 0: fused weights WT[m*DIM+d] = bf16(sum_j W[d,j]*proj[m,j]); bout[m]=b.proj[m] ----
__global__ void prep_w_kernel(const float* __restrict__ W, const float* __restrict__ b,
                              const float* __restrict__ proj,
                              unsigned short* __restrict__ WT, float* __restrict__ bout) {
  int t = blockIdx.x * blockDim.x + threadIdx.x;
  if (t < MF * DIM) {
    int m = t / DIM, d = t % DIM;
    const float* wr = W + (size_t)d * DIM;
    const float* pr = proj + (size_t)m * DIM;
    float acc = 0.f;
    for (int j = 0; j < DIM; ++j) acc = fmaf(wr[j], pr[j], acc);
    WT[t] = f2bf(acc);
  }
  if (t < MF) {
    const float* pr = proj + (size_t)t * DIM;
    float acc = 0.f;
    for (int j = 0; j < DIM; ++j) acc = fmaf(b[j], pr[j], acc);
    bout[t] = acc;
  }
}

// ---- Kernel 1a: x (f32) -> xb (bf16), vectorized 4 elems/thread ----
__global__ void conv_bf_kernel(const float* __restrict__ src, unsigned short* __restrict__ dst) {
  size_t t = (size_t)(blockIdx.x * blockDim.x + threadIdx.x) * 4;
  v4f v = *(const v4f*)(src + t);
  unsigned short h[4];
#pragma unroll
  for (int j = 0; j < 4; ++j) h[j] = f2bf(v[j]);
  *(uint64_t*)(dst + t) = ((uint64_t)h[3] << 48) | ((uint64_t)h[2] << 32) |
                          ((uint64_t)h[1] << 16) | (uint64_t)h[0];
}

// ---- Kernel 1b: chi (NxD f32) -> chiT (DxN bf16), LDS-tiled transpose ----
__global__ void transpose_bf_kernel(const float* __restrict__ src, unsigned short* __restrict__ dstT) {
  __shared__ float tile[32][33];
  int n0 = blockIdx.x * 32, d0 = blockIdx.y * 32;
  tile[threadIdx.y][threadIdx.x] = src[(size_t)(n0 + threadIdx.y) * DIM + (d0 + threadIdx.x)];
  __syncthreads();
  dstT[(size_t)(d0 + threadIdx.y) * N_PTS + (n0 + threadIdx.x)] = f2bf(tile[threadIdx.x][threadIdx.y]);
}

// ---- Kernel 2: phi = relu(xb @ fusedW + fusedB) + eps ; wave computes 16x64 tile ----
__global__ void phi_kernel(const unsigned short* __restrict__ xb, const unsigned short* __restrict__ WT,
                           const float* __restrict__ bproj,
                           unsigned short* __restrict__ outRM, unsigned short* __restrict__ outT) {
  int wid = (blockIdx.x * blockDim.x + threadIdx.x) >> 5;
  const int mBlk = MF / 64;                 // 8
  int tn = wid / mBlk, tm = wid % mBlk;
  int n0 = tn * 16, m0 = tm * 64;
  v8f acc[4] = {};
  for (int kk = 0; kk < DIM; kk += 32) {
    v16bf a = loadA_bf(xb, DIM, n0, kk);
#pragma unroll
    for (int j = 0; j < 4; ++j) {
      v16bf b = loadBT_bf(WT, DIM, m0 + 16 * j, kk);
      acc[j] = WMMA_BF16(a, b, acc[j]);
    }
  }
  int lane  = threadIdx.x & 31;
  int rbase = n0 + ((lane >> 4) << 3);
#pragma unroll
  for (int j = 0; j < 4; ++j) {
    int col = m0 + 16 * j + (lane & 15);
    float bias = bproj[col];
#pragma unroll
    for (int r = 0; r < 8; ++r) {
      float v = acc[j][r] + bias;
      v = (v > 0.f ? v : 0.f) + KEPS;
      unsigned short h = f2bf(v);
      int row = rbase + r;
      if (outRM) outRM[(size_t)row * MF + col] = h;
      if (outT)  outT[(size_t)col * N_PTS + row] = h;
    }
  }
}

// ---- Kernel 3: kv partials: kv[m,d] = sum_n k'[n,m]*chi[n,d] ; wave computes 32x32 tile ----
__global__ void kv_kernel(const unsigned short* __restrict__ kpT, const unsigned short* __restrict__ chiT,
                          float* __restrict__ kv_part) {
  int wid = (blockIdx.x * blockDim.x + threadIdx.x) >> 5;
  const int tiles = (MF / 32) * (DIM / 32);             // 256
  int c = wid / tiles;
  int t = wid % tiles;
  int td = t % (DIM / 32), tm = t / (DIM / 32);
  int m0 = tm * 32, d0 = td * 32;
  const int nb = N_PTS / KV_CHUNKS;
  int nstart = c * nb;
  v8f acc00 = {}, acc01 = {}, acc10 = {}, acc11 = {};
  for (int kk = 0; kk < nb; kk += 32) {
    v16bf a0 = loadA_bf(kpT, N_PTS, m0,      nstart + kk);
    v16bf a1 = loadA_bf(kpT, N_PTS, m0 + 16, nstart + kk);
    v16bf b0 = loadBT_bf(chiT, N_PTS, d0,      nstart + kk);
    v16bf b1 = loadBT_bf(chiT, N_PTS, d0 + 16, nstart + kk);
    acc00 = WMMA_BF16(a0, b0, acc00);
    acc01 = WMMA_BF16(a0, b1, acc01);
    acc10 = WMMA_BF16(a1, b0, acc10);
    acc11 = WMMA_BF16(a1, b1, acc11);
  }
  int lane  = threadIdx.x & 31;
  int dcol  = d0 + (lane & 15);
  int mbase = m0 + ((lane >> 4) << 3);
  float* dst = kv_part + (size_t)c * MF * DIM;
#pragma unroll
  for (int r = 0; r < 8; ++r) {
    dst[(size_t)(mbase + r) * DIM + dcol]           = acc00[r];
    dst[(size_t)(mbase + r) * DIM + dcol + 16]      = acc01[r];
    dst[(size_t)(mbase + 16 + r) * DIM + dcol]      = acc10[r];
    dst[(size_t)(mbase + 16 + r) * DIM + dcol + 16] = acc11[r];
  }
}

// ---- Kernel 4: reduce partials -> kvT bf16 (DxM, transposed for final GEMM B operand) ----
__global__ void kv_reduce_kernel(const float* __restrict__ kv_part, unsigned short* __restrict__ kvT) {
  int t = blockIdx.x * blockDim.x + threadIdx.x;
  if (t >= MF * DIM) return;
  float s = 0.f;
#pragma unroll
  for (int c = 0; c < KV_CHUNKS; ++c) s += kv_part[(size_t)c * MF * DIM + t];
  int m = t / DIM, d = t % DIM;
  kvT[(size_t)d * MF + m] = f2bf(s);
}

// ---- Kernel 5: ks_sum[m] = sum_n k'[n,m]  (reads contiguous row of kpT) ----
__global__ void ks_sum_kernel(const unsigned short* __restrict__ kpT, float* __restrict__ ks) {
  __shared__ float red[256];
  int m = blockIdx.x;
  const unsigned short* row = kpT + (size_t)m * N_PTS;
  float s = 0.f;
  for (int n = threadIdx.x; n < N_PTS; n += 256) s += bf2f(row[n]);
  red[threadIdx.x] = s;
  __syncthreads();
  for (int w = 128; w > 0; w >>= 1) {
    if (threadIdx.x < w) red[threadIdx.x] += red[threadIdx.x + w];
    __syncthreads();
  }
  if (threadIdx.x == 0) ks[m] = red[0];
}

// ---- Kernel 6: den[n] = q'[n,:].ks ; stabilize ; store reciprocal ----
__global__ void den_kernel(const unsigned short* __restrict__ qp, const float* __restrict__ ks,
                           float* __restrict__ inv_den) {
  int n = blockIdx.x * blockDim.x + threadIdx.x;
  if (n >= N_PTS) return;
  const unsigned short* row = qp + (size_t)n * MF;
  float acc = 0.f;
  for (int m = 0; m < MF; ++m) acc = fmaf(bf2f(row[m]), ks[m], acc);
  if (fabsf(acc) <= NSTAB) acc += 2.0f * NSTAB;
  inv_den[n] = 1.0f / acc;
}

// ---- Kernel 7: out = (q' @ kv) * inv_den[:,None] ; wave computes 16x64 tile ----
__global__ void out_kernel(const unsigned short* __restrict__ qp, const unsigned short* __restrict__ kvT,
                           const float* __restrict__ inv_den, float* __restrict__ out) {
  int wid = (blockIdx.x * blockDim.x + threadIdx.x) >> 5;
  const int dBlk = DIM / 64;                // 8
  int tn = wid / dBlk, td = wid % dBlk;
  int n0 = tn * 16, d0 = td * 64;
  v8f acc[4] = {};
  for (int kk = 0; kk < MF; kk += 32) {
    v16bf a = loadA_bf(qp, MF, n0, kk);
#pragma unroll
    for (int j = 0; j < 4; ++j) {
      v16bf b = loadBT_bf(kvT, MF, d0 + 16 * j, kk);
      acc[j] = WMMA_BF16(a, b, acc[j]);
    }
  }
  int lane  = threadIdx.x & 31;
  int rbase = n0 + ((lane >> 4) << 3);
#pragma unroll
  for (int r = 0; r < 8; ++r) {
    int row = rbase + r;
    float s = inv_den[row];
#pragma unroll
    for (int j = 0; j < 4; ++j) {
      int col = d0 + 16 * j + (lane & 15);
      out[(size_t)row * DIM + col] = acc[j][r] * s;
    }
  }
}

extern "C" void kernel_launch(void* const* d_in, const int* in_sizes, int n_in,
                              void* d_out, int out_size, void* d_ws, size_t ws_size,
                              hipStream_t stream) {
  const float* chi  = (const float*)d_in[0];
  const float* x    = (const float*)d_in[1];
  const float* Wq   = (const float*)d_in[2];
  const float* bq   = (const float*)d_in[3];
  const float* Wk   = (const float*)d_in[4];
  const float* bk   = (const float*)d_in[5];
  const float* proj = (const float*)d_in[6];
  float* out = (float*)d_out;
  (void)in_sizes; (void)n_in; (void)out_size; (void)ws_size;

  char* ws = (char*)d_ws;
  size_t off = 0;
  auto alloc = [&](size_t bytes) {
    char* p = ws + off;
    off = (off + bytes + 255) & ~(size_t)255;
    return p;
  };
  unsigned short* WqT  = (unsigned short*)alloc((size_t)MF * DIM * 2);
  unsigned short* WkT  = (unsigned short*)alloc((size_t)MF * DIM * 2);
  float*          bqp  = (float*)alloc(MF * 4);
  float*          bkp  = (float*)alloc(MF * 4);
  unsigned short* xb   = (unsigned short*)alloc((size_t)N_PTS * DIM * 2);
  unsigned short* chiT = (unsigned short*)alloc((size_t)DIM * N_PTS * 2);
  unsigned short* qp   = (unsigned short*)alloc((size_t)N_PTS * MF * 2);
  unsigned short* kpT  = (unsigned short*)alloc((size_t)MF * N_PTS * 2);
  float*          kvp  = (float*)alloc((size_t)KV_CHUNKS * MF * DIM * 4);
  unsigned short* kvT  = (unsigned short*)alloc((size_t)DIM * MF * 2);
  float*          ks   = (float*)alloc(MF * 4);
  float*          invd = (float*)alloc((size_t)N_PTS * 4);

  dim3 b256(256);
  prep_w_kernel<<<(MF * DIM + 255) / 256, b256, 0, stream>>>(Wq, bq, proj, WqT, bqp);
  prep_w_kernel<<<(MF * DIM + 255) / 256, b256, 0, stream>>>(Wk, bk, proj, WkT, bkp);
  conv_bf_kernel<<<((size_t)N_PTS * DIM / 4 + 255) / 256, b256, 0, stream>>>(x, xb);
  transpose_bf_kernel<<<dim3(N_PTS / 32, DIM / 32), dim3(32, 32), 0, stream>>>(chi, chiT);

  int phiWaves = (N_PTS / 16) * (MF / 64);               // 16384 waves
  phi_kernel<<<phiWaves / 8, b256, 0, stream>>>(xb, WqT, bqp, qp, (unsigned short*)nullptr);
  phi_kernel<<<phiWaves / 8, b256, 0, stream>>>(xb, WkT, bkp, (unsigned short*)nullptr, kpT);

  int kvWaves = KV_CHUNKS * (MF / 32) * (DIM / 32);      // 2048 waves
  kv_kernel<<<kvWaves / 8, b256, 0, stream>>>(kpT, chiT, kvp);
  kv_reduce_kernel<<<(MF * DIM + 255) / 256, b256, 0, stream>>>(kvp, kvT);

  ks_sum_kernel<<<MF, b256, 0, stream>>>(kpT, ks);
  den_kernel<<<(N_PTS + 255) / 256, b256, 0, stream>>>(qp, ks, invd);

  int outWaves = (N_PTS / 16) * (DIM / 64);              // 16384 waves
  out_kernel<<<outWaves / 8, b256, 0, stream>>>(qp, kvT, invd, out);
}
